// VQVAELayer_20684562497845
// MI455X (gfx1250) — matrix operand order repaired and checked by reading, not compiled, not run on hardware
//
#include <hip/hip_runtime.h>

// ---------------------------------------------------------------------------
// VQ-VAE codebook lookup + EMA update for MI455X (gfx1250, wave32, WMMA).
//   scores[n,k] = ||vq_k||^2 - 2 * <x_n, vq_k>    (||x||^2 dropped: argmin-inv.)
//   dot products via v_wmma_f32_16x16x32_bf16 (codebook + x tiles in LDS bf16)
//   ||vq||^2 kept in f32. Segment sums via global_atomic_add_f32 into
//   pre-scaled EMA outputs.
// ---------------------------------------------------------------------------

typedef __attribute__((ext_vector_type(16))) __bf16 v16bf;
typedef __attribute__((ext_vector_type(8)))  __bf16 v8bf;
typedef __attribute__((ext_vector_type(4)))  __bf16 v4bf;
typedef __attribute__((ext_vector_type(8)))  float  v8f;

#define N_TOK 262144
#define DIM   64
#define KCB   512
#define GAMMA 0.99f
#define ROWS_PER_WG 128   // 8 waves * 16 rows

// ---------------------------------------------------------------------------
// K1: build bf16 codebook + ||vq||^2 in workspace; init EMA accumulators:
//     out_cs = gamma * centroid_sum ; out_cn = gamma * centroid_n
// ---------------------------------------------------------------------------
__global__ void k_prep(const float* __restrict__ vq,
                       const float* __restrict__ cs_in,
                       const float* __restrict__ cn_in,
                       __bf16* __restrict__ vq_bf,
                       float*  __restrict__ vqn2,
                       float*  __restrict__ out_cs,
                       float*  __restrict__ out_cn) {
  int t = blockIdx.x * blockDim.x + threadIdx.x;   // 0 .. K*D-1
  if (t >= KCB * DIM) return;
  float v = vq[t];
  vq_bf[t]  = (__bf16)v;
  out_cs[t] = GAMMA * cs_in[t];
  int k = t >> 6;
  if ((t & 63) == 0) {
    float s = 0.f;
#pragma unroll
    for (int j = 0; j < DIM; ++j) { float w = vq[k * DIM + j]; s = fmaf(w, w, s); }
    vqn2[k]   = s;
    out_cn[k] = GAMMA * cn_in[k];
  }
}

// ---------------------------------------------------------------------------
// K2: main kernel. Per workgroup: 128 rows of x. Per wave: 16 rows vs all
// 512 codewords = 32 tiles of 16x16, each tile = 2x wmma_f32_16x16x32_bf16.
// ---------------------------------------------------------------------------
__launch_bounds__(256)
__global__ void k_vq_main(const float* __restrict__ x,
                          const float* __restrict__ vq_f32,
                          const __bf16* __restrict__ vq_bf,
                          const float* __restrict__ vqn2,
                          float* __restrict__ out_q,
                          float* __restrict__ out_cs,
                          float* __restrict__ out_cn) {
  extern __shared__ char smem[];
  __bf16* lds_vq  = (__bf16*)smem;                     // 512*64*2 = 65536 B
  float*  lds_n2  = (float*)(smem + 65536);            //  512*4   =  2048 B
  __bf16* lds_x   = (__bf16*)(smem + 67584);           // 128*64*2 = 16384 B
  int*    lds_idx = (int*)(smem + 67584 + 16384);      //  128*4   =   512 B

  const int tid   = threadIdx.x;
  const int wgRow = blockIdx.x * ROWS_PER_WG;

  // ---- stage bf16 codebook + ||v||^2 into LDS (L2-hot after k_prep) ----
  {
    const uint4* src = (const uint4*)vq_bf;            // 4096 x 16B
    uint4*       dst = (uint4*)lds_vq;
#pragma unroll
    for (int i = 0; i < 16; ++i) dst[tid + i * 256] = src[tid + i * 256];
    lds_n2[tid]       = vqn2[tid];
    lds_n2[tid + 256] = vqn2[tid + 256];
  }
  // ---- stage this WG's x tile as bf16 (vectorized f32x4 -> bf16x4) ----
  {
    const float4* xs = (const float4*)(x + (size_t)wgRow * DIM);  // 2048 x 16B
#pragma unroll
    for (int i = 0; i < 8; ++i) {
      float4 f = xs[tid + i * 256];
      v4bf b = { (__bf16)f.x, (__bf16)f.y, (__bf16)f.z, (__bf16)f.w };
      *(v4bf*)(lds_x + (size_t)(tid + i * 256) * 4) = b;
    }
  }
  __syncthreads();

  const int lane = tid & 31;
  const int wave = tid >> 5;
  const int half = (lane >> 4) & 1;   // which 16-lane half
  const int lr   = lane & 15;

  // ---- A fragments (16x32 bf16): lane holds row lr,
  //      elems 0..15 = K base+half*16.. (contiguous) ----
  const __bf16* xa = lds_x + (wave * 16 + lr) * DIM;
  v16bf a0, a1;
  {
    union { v16bf v; struct { v8bf lo; v8bf hi; } p; } ua;
    ua.p.lo = *(const v8bf*)(xa + 0  + half * 16);
    ua.p.hi = *(const v8bf*)(xa + 8  + half * 16);
    a0 = ua.v;
    ua.p.lo = *(const v8bf*)(xa + 32 + half * 16);
    ua.p.hi = *(const v8bf*)(xa + 40 + half * 16);
    a1 = ua.v;
  }

  float minv[8];
  int   mini[8];
#pragma unroll
  for (int v = 0; v < 8; ++v) { minv[v] = 3.4e38f; mini[v] = 0; }

  // ---- sweep 32 codeword tiles: 2 WMMAs + fused fma/min epilogue ----
#pragma unroll 4
  for (int t = 0; t < 32; ++t) {
    const int c0 = t * 16;
    // B fragment (32x16 bf16): lane = column (codeword c0+lr),
    // 16 contiguous K values starting at ks*32 + half*16.
    const __bf16* brow = lds_vq + (c0 + lr) * DIM + half * 16;
    v16bf b0 = *(const v16bf*)(brow);
    v16bf b1 = *(const v16bf*)(brow + 32);
    v8f c = {};
    c = __builtin_amdgcn_wmma_f32_16x16x32_bf16(false, a0, false, b0,
                                                (short)0, c, false, false);
    c = __builtin_amdgcn_wmma_f32_16x16x32_bf16(false, a1, false, b1,
                                                (short)0, c, false, false);
    const float n2 = lds_n2[c0 + lr];
    const int   n  = c0 + lr;
#pragma unroll
    for (int v = 0; v < 8; ++v) {
      float s = fmaf(c[v], -2.0f, n2);     // ||v||^2 - 2<x,v>
      if (s < minv[v]) { minv[v] = s; mini[v] = n; }  // strict <: first-min ties
    }
  }

  // ---- per-row argmin across the 16 lanes of each half-wave ----
#pragma unroll
  for (int v = 0; v < 8; ++v) {
    float mv = minv[v];
    int   mi = mini[v];
#pragma unroll
    for (int off = 8; off > 0; off >>= 1) {
      float ov = __shfl_xor(mv, off, 32);
      int   oi = __shfl_xor(mi, off, 32);
      if (ov < mv || (ov == mv && oi < mi)) { mv = ov; mi = oi; }
    }
    if (lr == v) lds_idx[wave * 16 + v + half * 8] = mi;  // row M = v + 8*half
  }
  __syncthreads();

  // ---- epilogue: gather quantized rows, atomic EMA scatter ----
  const float alpha = 1.0f - GAMMA;
#pragma unroll 4
  for (int i = 0; i < 32; ++i) {
    int e   = tid + i * 256;           // 0..8191 over 128 rows x 64 dims
    int row = e >> 6;
    int d   = e & 63;
    int k   = lds_idx[row];
    size_t g = (size_t)wgRow * DIM + e;
    __builtin_nontemporal_store(vq_f32[k * DIM + d], &out_q[g]);  // streaming 64MB
    atomicAdd(&out_cs[k * DIM + d], alpha * x[g]);
    if (d == 0) atomicAdd(&out_cn[k], alpha);
  }
}

// ---------------------------------------------------------------------------
// K3: new_vq = new_centroid_sum / new_centroid_n  (centroid_n >= gamma > 0)
// ---------------------------------------------------------------------------
__global__ void k_final(const float* __restrict__ cs,
                        const float* __restrict__ cn,
                        float* __restrict__ out_vq) {
  int t = blockIdx.x * blockDim.x + threadIdx.x;
  if (t < KCB * DIM) out_vq[t] = cs[t] / cn[t >> 6];
}

// ---------------------------------------------------------------------------
extern "C" void kernel_launch(void* const* d_in, const int* in_sizes, int n_in,
                              void* d_out, int out_size, void* d_ws, size_t ws_size,
                              hipStream_t stream) {
  const float* x   = (const float*)d_in[0];
  const float* vq  = (const float*)d_in[1];
  const float* cs0 = (const float*)d_in[2];
  const float* cn0 = (const float*)d_in[3];

  float* out    = (float*)d_out;
  float* out_q  = out;                            // [N*D]
  float* out_vq = out + (size_t)N_TOK * DIM;      // [K*D]
  float* out_cs = out_vq + KCB * DIM;             // [K*D]
  float* out_cn = out_cs + KCB * DIM;             // [K]

  __bf16* vq_bf = (__bf16*)d_ws;                                  // 64 KB
  float*  vqn2  = (float*)((char*)d_ws + (size_t)KCB * DIM * 2);  // 2 KB
  (void)in_sizes; (void)n_in; (void)out_size; (void)ws_size;

  k_prep<<<(KCB * DIM) / 256, 256, 0, stream>>>(vq, cs0, cn0, vq_bf, vqn2,
                                                out_cs, out_cn);

  const size_t smem = 65536 + 2048 + 16384 + 512;  // 82.5 KB (< 320KB/WGP)
  hipFuncSetAttribute((const void*)k_vq_main,
                      hipFuncAttributeMaxDynamicSharedMemorySize, (int)smem);
  k_vq_main<<<N_TOK / ROWS_PER_WG, 256, smem, stream>>>(x, vq, vq_bf, vqn2,
                                                        out_q, out_cs, out_cn);

  k_final<<<(KCB * DIM) / 256, 256, 0, stream>>>(out_cs, out_cn, out_vq);
}
